// CSMAdapter_70042326664019
// MI455X (gfx1250) — compile-verified, matrix operand order checked by reading, not run on hardware
//
#include <hip/hip_runtime.h>
#include <hip/hip_bf16.h>

// ---------------------------------------------------------------------------
// CDNA5 (gfx1250) implementation of the CSM adapter forward pass.
// wave32, WMMA 16x16x32 bf16 -> f32 accumulate.
// ---------------------------------------------------------------------------

typedef __bf16 bf16_t;
typedef __attribute__((ext_vector_type(16))) __bf16 v16bf;
typedef __attribute__((ext_vector_type(8)))  float  v8f;

union Frag16 { v16bf v; uint4 q[2]; };

__device__ inline bf16_t f2bf(float f) {
    union { float f; unsigned u; } a; a.f = f;
    unsigned r = a.u + 0x7FFFu + ((a.u >> 16) & 1u);   // RNE
    union { unsigned short s; bf16_t b; } o; o.s = (unsigned short)(r >> 16);
    return o.b;
}
__device__ inline float bf2f(bf16_t b) {
    union { unsigned short s; bf16_t b; } i; i.b = b;
    union { unsigned u; float f; } o; o.u = ((unsigned)i.s) << 16;
    return o.f;
}
__device__ inline bf16_t bf_zero() {
    union { unsigned short s; bf16_t b; } o; o.s = 0; return o.b;
}
__device__ inline v8f vzero8() {
    v8f z;
#pragma unroll
    for (int i = 0; i < 8; ++i) z[i] = 0.f;
    return z;
}
// Load one 16-bit WMMA A/B fragment slice: 8 contiguous bf16 at p (K=kb..kb+7)
// and 8 contiguous bf16 at p+16 (K=kb+16..kb+23).  Caller picks p per lane.
__device__ inline v16bf load_frag(const bf16_t* p) {
    Frag16 u;
    u.q[0] = *(const uint4*)(p);
    u.q[1] = *(const uint4*)(p + 16);
    return u.v;
}
__device__ inline v8f wmma_bf16(v16bf a, v16bf b, v8f c) {
    return __builtin_amdgcn_wmma_f32_16x16x32_bf16(
        false, a, false, b, (short)0, c, false, false);
}
__device__ inline float gelu_tanh(float x) {
    float x3 = x * x * x;
    return 0.5f * x * (1.f + tanhf(0.7978845608f * (x + 0.044715f * x3)));
}
__device__ inline float silu_f(float x) { return x / (1.f + __expf(-x)); }

// ---------------------------------------------------------------------------
// GEMM: C[M,N] = act(A[M,K] @ B^T + bias) * gate + residual
// A is bf16 row-major [M,K].
// B_KN == false: B is bf16 [N,K] row-major (i.e. torch Linear weight).
// B_KN == true : B is bf16 [K,N] row-major (transposed during LDS staging).
// ---------------------------------------------------------------------------
template<bool B_KN, bool OUT_BF16>
__global__ __launch_bounds__(256)
void k_gemm(const bf16_t* __restrict__ A, const bf16_t* __restrict__ Bm,
            void* __restrict__ Cout, int M, int N, int K,
            const float* __restrict__ bias, int act,
            const float* __restrict__ gate, int gateStride, int gateGroup,
            const float* __restrict__ residual)
{
    __shared__ bf16_t sA[128][32];
    __shared__ bf16_t sB[128][32];   // always stored [n][k]

    const int tid  = threadIdx.x;
    const int wave = tid >> 5;
    const int lane = tid & 31;
    const int l15  = lane & 15;
    const int lh   = lane >> 4;           // lane half
    const int n0   = blockIdx.x * 128;
    const int m0   = blockIdx.y * 128;
    const int wm   = wave & 1;            // 2 waves along M (64 rows each)
    const int wn   = wave >> 1;           // 4 waves along N (32 cols each)

    v8f acc[4][2];
#pragma unroll
    for (int mi = 0; mi < 4; ++mi)
#pragma unroll
        for (int ni = 0; ni < 2; ++ni) acc[mi][ni] = vzero8();

    const int ar   = tid >> 1;            // staging row 0..127
    const int aseg = (tid & 1) * 16;      // staging half-row segment

    for (int k0 = 0; k0 < K; k0 += 32) {
        // ---- stage A tile ----
        {
            int gr = m0 + ar;
            uint4 z0 = make_uint4(0u, 0u, 0u, 0u), z1 = z0;
            if (gr < M) {
                const uint4* p = (const uint4*)(A + (size_t)gr * K + k0 + aseg);
                z0 = p[0]; z1 = p[1];
                if (k0 + 32 < K)  // hint next K tile into cache (global_prefetch_b8)
                    __builtin_prefetch(A + (size_t)gr * K + k0 + 32 + aseg, 0, 1);
            }
            *(uint4*)&sA[ar][aseg]     = z0;
            *(uint4*)&sA[ar][aseg + 8] = z1;
        }
        // ---- stage B tile ----
        if (!B_KN) {
            int gr = n0 + ar;
            uint4 z0 = make_uint4(0u, 0u, 0u, 0u), z1 = z0;
            if (gr < N) {
                const uint4* p = (const uint4*)(Bm + (size_t)gr * K + k0 + aseg);
                z0 = p[0]; z1 = p[1];
                if (k0 + 32 < K)
                    __builtin_prefetch(Bm + (size_t)gr * K + k0 + 32 + aseg, 0, 1);
            }
            *(uint4*)&sB[ar][aseg]     = z0;
            *(uint4*)&sB[ar][aseg + 8] = z1;
        } else {
#pragma unroll
            for (int it = 0; it < 16; ++it) {
                int idx = tid + it * 256;
                int n = idx & 127, k = idx >> 7;
                bf16_t v = bf_zero();
                if (n0 + n < N) v = Bm[(size_t)(k0 + k) * N + n0 + n];
                sB[n][k] = v;
            }
        }
        __syncthreads();

        // ---- compute: one WMMA K-step (BK == 32) ----
        v16bf bfr[2];
        bfr[0] = load_frag(&sB[wn * 32 + 0 + l15][lh * 8]);
        bfr[1] = load_frag(&sB[wn * 32 + 16 + l15][lh * 8]);
#pragma unroll
        for (int mi = 0; mi < 4; ++mi) {
            v16bf afr = load_frag(&sA[wm * 64 + mi * 16 + l15][lh * 8]);
            acc[mi][0] = wmma_bf16(afr, bfr[0], acc[mi][0]);
            acc[mi][1] = wmma_bf16(afr, bfr[1], acc[mi][1]);
        }
        __syncthreads();
    }

    // ---- epilogue ----
#pragma unroll
    for (int mi = 0; mi < 4; ++mi) {
#pragma unroll
        for (int ni = 0; ni < 2; ++ni) {
            int gn = n0 + wn * 32 + ni * 16 + l15;
            if (gn >= N) continue;
            int gm0 = m0 + wm * 64 + mi * 16;
#pragma unroll
            for (int v = 0; v < 8; ++v) {
                int gm = gm0 + v + 8 * lh;
                if (gm >= M) continue;
                float val = acc[mi][ni][v];
                if (bias) val += bias[gn];
                if (act == 1)      val = gelu_tanh(val);
                else if (act == 2) val = silu_f(val);
                if (gate) val *= gate[(size_t)(gm / gateGroup) * gateStride + gn];
                if (residual) val += residual[(size_t)gm * N + gn];
                if (OUT_BF16) ((bf16_t*)Cout)[(size_t)gm * N + gn] = f2bf(val);
                else          ((float*)Cout)[(size_t)gm * N + gn] = val;
            }
        }
    }
}

// ---------------------------------------------------------------------------
// Flash attention, per-(b,h,64-query block), 4 waves x 16 query rows.
// Q,K,V,O are bf16 [B*T, 1024] (head h at column h*64). WMMA for QK^T and PV.
// ---------------------------------------------------------------------------
__global__ __launch_bounds__(128)
void k_flash_attn(const bf16_t* __restrict__ Q, const bf16_t* __restrict__ Kd,
                  const bf16_t* __restrict__ V, const unsigned char* __restrict__ mask,
                  bf16_t* __restrict__ O, int Bsz, int H, int T, int Dm)
{
    const int DH = 64;
    __shared__ bf16_t sVt[64][64];      // V^T tile: [d][key]
    __shared__ bf16_t sP[4][16][64];    // per-wave probs for A-fragment reload

    const int tid  = threadIdx.x;
    const int wave = tid >> 5;
    const int lane = tid & 31;
    const int l15  = lane & 15;
    const int lh   = lane >> 4;
    const int b = blockIdx.z, h = blockIdx.y;
    const int q0 = blockIdx.x * 64 + wave * 16;

    // Q fragments for K-dim 0..63 (two 16x32 fragments), straight from global.
    size_t qrow = (size_t)(b * T + q0 + l15) * Dm + h * DH;
    v16bf qf[2];
    qf[0] = load_frag(Q + qrow + lh * 8);
    qf[1] = load_frag(Q + qrow + 32 + lh * 8);

    float rmax[8], rsum[8];
    v8f o[4];
#pragma unroll
    for (int e = 0; e < 8; ++e) { rmax[e] = -3.0e38f; rsum[e] = 0.f; }
#pragma unroll
    for (int d = 0; d < 4; ++d) o[d] = vzero8();

    for (int j0 = 0; j0 < T; j0 += 64) {
        __syncthreads();     // previous iteration done with sVt
        // stage V^T tile cooperatively
        for (int idx = tid; idx < 64 * 64; idx += 128) {
            int kk = idx >> 6, d = idx & 63;
            sVt[d][kk] = V[(size_t)(b * T + j0 + kk) * Dm + h * DH + d];
        }

        // ---- scores S = Q @ K^T (64 keys) ----
        v8f s[4];
#pragma unroll
        for (int nt = 0; nt < 4; ++nt) s[nt] = vzero8();
#pragma unroll
        for (int f = 0; f < 2; ++f) {
#pragma unroll
            for (int nt = 0; nt < 4; ++nt) {
                size_t krow = (size_t)(b * T + j0 + nt * 16 + l15) * Dm
                              + h * DH + f * 32 + lh * 8;
                v16bf kf = load_frag(Kd + krow);
                s[nt] = wmma_bf16(qf[f], kf, s[nt]);
            }
        }
        // scale + key mask (column == lane&15 in C layout)
#pragma unroll
        for (int nt = 0; nt < 4; ++nt) {
            int kcol = j0 + nt * 16 + l15;
            bool keep = mask[(size_t)b * T + kcol] != 0;
#pragma unroll
            for (int e = 0; e < 8; ++e) {
                float sv = s[nt][e] * 0.125f;
                s[nt][e] = keep ? sv : -3.0e38f;
            }
        }
        // ---- online softmax: row stats via 16-lane xor reductions ----
        float bm[8];
#pragma unroll
        for (int e = 0; e < 8; ++e) {
            float m = fmaxf(fmaxf(s[0][e], s[1][e]), fmaxf(s[2][e], s[3][e]));
#pragma unroll
            for (int off = 1; off < 16; off <<= 1)
                m = fmaxf(m, __shfl_xor(m, off, 32));
            bm[e] = m;
        }
        float alpha[8], nm[8], ps[8];
#pragma unroll
        for (int e = 0; e < 8; ++e) {
            nm[e] = fmaxf(rmax[e], bm[e]);
            alpha[e] = __expf(rmax[e] - nm[e]);
            ps[e] = 0.f;
        }
#pragma unroll
        for (int nt = 0; nt < 4; ++nt) {
#pragma unroll
            for (int e = 0; e < 8; ++e) {
                float p = __expf(s[nt][e] - nm[e]);
                ps[e] += p;
                sP[wave][e + 8 * lh][nt * 16 + l15] = f2bf(p);  // transpose via LDS
            }
        }
#pragma unroll
        for (int e = 0; e < 8; ++e) {
            float t = ps[e];
#pragma unroll
            for (int off = 1; off < 16; off <<= 1)
                t += __shfl_xor(t, off, 32);
            rsum[e] = rsum[e] * alpha[e] + t;
            rmax[e] = nm[e];
        }
#pragma unroll
        for (int d = 0; d < 4; ++d)
#pragma unroll
            for (int e = 0; e < 8; ++e) o[d][e] *= alpha[e];

        __syncthreads();     // sVt visible, sP written

        // ---- O += P @ V ----
#pragma unroll
        for (int f = 0; f < 2; ++f) {
            v16bf pf = load_frag(&sP[wave][l15][f * 32 + lh * 8]);
#pragma unroll
            for (int d = 0; d < 4; ++d) {
                v16bf vf = load_frag(&sVt[d * 16 + l15][f * 32 + lh * 8]);
                o[d] = wmma_bf16(pf, vf, o[d]);
            }
        }
    }

    // ---- write O / rowsum ----
#pragma unroll
    for (int d = 0; d < 4; ++d) {
        int col = h * DH + d * 16 + l15;
#pragma unroll
        for (int v = 0; v < 8; ++v) {
            int grow = b * T + q0 + v + 8 * lh;
            O[(size_t)grow * Dm + col] = f2bf(o[d][v] / fmaxf(rsum[v], 1e-20f));
        }
    }
}

// ---------------------------------------------------------------------------
// Small / elementwise kernels
// ---------------------------------------------------------------------------
__global__ void k_time_embed(const float* __restrict__ ts, float* __restrict__ out)
{
    int b = blockIdx.x, i = threadIdx.x;     // 256 threads
    float t = ts[b] * 1000.f;
    if (i < 128) {
        float f = __expf((float)i * (-9.210340372f / 127.f));
        out[b * 256 + i] = __sinf(t * f);
    } else {
        float f = __expf((float)(i - 128) * (-9.210340372f / 127.f));
        out[b * 256 + i] = __cosf(t * f);
    }
}

__global__ void k_small_linear(const float* __restrict__ X, const float* __restrict__ W,
                               const float* __restrict__ bias, float* __restrict__ Y,
                               int Br, int K, int N, int act)
{
    int i = blockIdx.x * blockDim.x + threadIdx.x;
    if (i >= Br * N) return;
    int b = i / N, n = i - b * N;
    const float4* x = (const float4*)(X + (size_t)b * K);
    const float4* w = (const float4*)(W + (size_t)n * K);
    float acc = 0.f;
    for (int k = 0; k < K / 4; ++k) {
        float4 a = x[k], c = w[k];
        acc += a.x * c.x + a.y * c.y + a.z * c.z + a.w * c.w;
    }
    if (bias) acc += bias[n];
    if (act == 2) acc = silu_f(acc);
    Y[i] = acc;
}

__global__ void k_silu(const float* __restrict__ in, float* __restrict__ out, int n)
{
    int i = blockIdx.x * blockDim.x + threadIdx.x;
    if (i < n) out[i] = silu_f(in[i]);
}

__global__ void k_f2bf(const float* __restrict__ in, bf16_t* __restrict__ out, int n)
{
    int i = blockIdx.x * blockDim.x + threadIdx.x;
    if (i < n) out[i] = f2bf(in[i]);
}

__global__ void k_transpose_f2bf(const float* __restrict__ in, bf16_t* __restrict__ out,
                                 int R, int C)
{
    int i = blockIdx.x * blockDim.x + threadIdx.x;
    if (i >= R * C) return;
    int r = i / C, c = i - r * C;
    out[(size_t)c * R + r] = f2bf(in[i]);
}

__global__ void k_mask_sigmoid(const bf16_t* __restrict__ wp, const float* __restrict__ sm,
                               bf16_t* __restrict__ out, int n)
{
    int i = blockIdx.x * blockDim.x + threadIdx.x;
    if (i < n) out[i] = f2bf(bf2f(wp[i]) * (1.f / (1.f + __expf(-sm[i]))));
}

// LayerNorm with optional per-batch AdaLN modulation or per-feature affine.
__global__ __launch_bounds__(256)
void k_ln(const float* __restrict__ X, int D,
          const float* __restrict__ shift, const float* __restrict__ scale,
          int modStride, int rowsPerB,
          const float* __restrict__ gamma, const float* __restrict__ beta,
          float* __restrict__ outF, bf16_t* __restrict__ outB)
{
    __shared__ float red[256];
    int row = blockIdx.x, tid = threadIdx.x;
    const float* x = X + (size_t)row * D;

    float s = 0.f;
    for (int i = tid; i < D; i += 256) s += x[i];
    red[tid] = s; __syncthreads();
    for (int off = 128; off > 0; off >>= 1) {
        if (tid < off) red[tid] += red[tid + off];
        __syncthreads();
    }
    float mu = red[0] / (float)D; __syncthreads();

    float s2 = 0.f;
    for (int i = tid; i < D; i += 256) { float d = x[i] - mu; s2 += d * d; }
    red[tid] = s2; __syncthreads();
    for (int off = 128; off > 0; off >>= 1) {
        if (tid < off) red[tid] += red[tid + off];
        __syncthreads();
    }
    float inv = rsqrtf(red[0] / (float)D + 1e-5f);
    int b = row / rowsPerB;

    for (int i = tid; i < D; i += 256) {
        float v = (x[i] - mu) * inv;
        if (scale) v = v * (1.f + scale[(size_t)b * modStride + i])
                      + shift[(size_t)b * modStride + i];
        if (gamma) v = v * gamma[i] + beta[i];
        if (outF) outF[(size_t)row * D + i] = v;
        if (outB) outB[(size_t)row * D + i] = f2bf(v);
    }
}

// Grouped Conv1d (kernel 3, pad 1, groups 16) over x stored [B,T,D].
__global__ void k_gconv(const float* __restrict__ X, const float* __restrict__ W,
                        const float* __restrict__ bias, float* __restrict__ Y,
                        int Bsz, int T, int D, int Cg, int act)
{
    int idx = blockIdx.x * blockDim.x + threadIdx.x;
    if (idx >= Bsz * T * D) return;
    int o = idx % D;
    int t = (idx / D) % T;
    int b = idx / (D * T);
    int cbase = (o / Cg) * Cg;
    const float* w = W + (size_t)o * Cg * 3;
    float acc = bias[o];
#pragma unroll
    for (int kk = 0; kk < 3; ++kk) {
        int tt = t + kk - 1;
        if (tt < 0 || tt >= T) continue;
        const float* xr = X + ((size_t)(b * T + tt)) * D + cbase;
        for (int i = 0; i < Cg; ++i) acc += w[i * 3 + kk] * xr[i];
    }
    if (act == 1) acc = gelu_tanh(acc);
    Y[idx] = acc;
}

// [B*T, NM] -> [B, NM, T]
__global__ void k_mel_out(const float* __restrict__ in, float* __restrict__ out,
                          int Bsz, int T, int Nm)
{
    int idx = blockIdx.x * blockDim.x + threadIdx.x;
    if (idx >= Bsz * Nm * T) return;
    int t = idx % T;
    int n = (idx / T) % Nm;
    int b = idx / (T * Nm);
    out[idx] = in[((size_t)(b * T + t)) * Nm + n];
}

// ---------------------------------------------------------------------------
// Host orchestration
// ---------------------------------------------------------------------------
static void gemm(hipStream_t st, const bf16_t* A, const bf16_t* B, bool bkn,
                 void* C, bool outBf, int M, int N, int K,
                 const float* bias, int act,
                 const float* gate, int gateStride, int gateGroup,
                 const float* residual)
{
    dim3 g((N + 127) / 128, (M + 127) / 128), blk(256);
    if (bkn) {
        if (outBf) k_gemm<true,  true ><<<g, blk, 0, st>>>(A, B, C, M, N, K, bias, act, gate, gateStride, gateGroup, residual);
        else       k_gemm<true,  false><<<g, blk, 0, st>>>(A, B, C, M, N, K, bias, act, gate, gateStride, gateGroup, residual);
    } else {
        if (outBf) k_gemm<false, true ><<<g, blk, 0, st>>>(A, B, C, M, N, K, bias, act, gate, gateStride, gateGroup, residual);
        else       k_gemm<false, false><<<g, blk, 0, st>>>(A, B, C, M, N, K, bias, act, gate, gateStride, gateGroup, residual);
    }
}

static inline dim3 g1(int n) { return dim3((n + 255) / 256); }

extern "C" void kernel_launch(void* const* d_in, const int* in_sizes, int n_in,
                              void* d_out, int out_size, void* d_ws, size_t ws_size,
                              hipStream_t stream)
{
    (void)in_sizes; (void)n_in; (void)out_size; (void)ws_size;
    const int B = 4, T = 512, D = 1024, DL = 3072, H = 8, NM = 100, FF = 4096;
    const int M2 = B * T;                 // 2048 rows of activations

    const float* llama   = (const float*)d_in[0];
    const float* tsteps  = (const float*)d_in[1];
    const unsigned char* maskp = (const unsigned char*)d_in[2];
    const float* P       = (const float*)d_in[3];
    const float* smask   = (const float*)d_in[4];
    const float* pdw     = (const float*)d_in[5];
    const float* pdb     = (const float*)d_in[6];
    const float* ipw     = (const float*)d_in[7];
    const float* ipb     = (const float*)d_in[8];
    const float* t_w1    = (const float*)d_in[9];
    const float* t_b1    = (const float*)d_in[10];
    const float* t_w2    = (const float*)d_in[11];
    const float* t_b2    = (const float*)d_in[12];
    const float* pc1_w   = (const float*)d_in[13];
    const float* pc1_b   = (const float*)d_in[14];
    const float* pc2_w   = (const float*)d_in[15];
    const float* pc2_b   = (const float*)d_in[16];
    const int BLK0 = 17;                  // 8 blocks x 16 tensors
    const float* fn_g    = (const float*)d_in[BLK0 + 8 * 16 + 0];
    const float* fn_b    = (const float*)d_in[BLK0 + 8 * 16 + 1];
    const float* mel_w   = (const float*)d_in[BLK0 + 8 * 16 + 2];
    const float* mel_b   = (const float*)d_in[BLK0 + 8 * 16 + 3];

    // ---- workspace carve-out ----
    char* wp = (char*)d_ws;
    auto alloc = [&](size_t bytes) -> void* {
        void* p = (void*)wp;
        wp += (bytes + 255) & ~(size_t)255;
        return p;
    };
    float*  sinEmb = (float*)alloc(B * 256 * 4);
    float*  temb1  = (float*)alloc(B * D * 4);
    float*  temb   = (float*)alloc(B * D * 4);
    float*  stemb  = (float*)alloc(B * D * 4);
    float*  m3     = (float*)alloc(B * 3 * D * 4);
    float*  m6     = (float*)alloc(B * 6 * D * 4);
    bf16_t* Pb     = (bf16_t*)alloc((size_t)D * D * 2);
    bf16_t* PTb    = (bf16_t*)alloc((size_t)D * D * 2);
    bf16_t* ipwb   = (bf16_t*)alloc((size_t)D * DL * 2);
    bf16_t* pdwb   = (bf16_t*)alloc((size_t)D * DL * 2);
    bf16_t* wprojb = (bf16_t*)alloc((size_t)D * D * 2);
    bf16_t* t1b    = (bf16_t*)alloc((size_t)D * D * 2);
    bf16_t* wprimb = (bf16_t*)alloc((size_t)D * D * 2);
    bf16_t* mwb    = (bf16_t*)alloc((size_t)D * D * 2);
    bf16_t* t2b    = (bf16_t*)alloc((size_t)D * D * 2);
    bf16_t* Mb     = (bf16_t*)alloc((size_t)D * D * 2);
    bf16_t* xinb   = (bf16_t*)alloc((size_t)M2 * DL * 2);
    bf16_t* x0b    = (bf16_t*)alloc((size_t)M2 * D * 2);
    float*  X      = (float*)alloc((size_t)M2 * D * 4);
    float*  Y1     = (float*)alloc((size_t)M2 * D * 4);
    bf16_t* hb     = (bf16_t*)alloc((size_t)M2 * D * 2);
    bf16_t* qb     = (bf16_t*)alloc((size_t)M2 * D * 2);
    bf16_t* kb     = (bf16_t*)alloc((size_t)M2 * D * 2);
    bf16_t* vb     = (bf16_t*)alloc((size_t)M2 * D * 2);
    bf16_t* attnb  = (bf16_t*)alloc((size_t)M2 * D * 2);
    bf16_t* fbuf   = (bf16_t*)alloc((size_t)M2 * FF * 2);
    bf16_t* wstage = (bf16_t*)alloc((size_t)FF * D * 2);
    float*  melTmp = (float*)alloc((size_t)M2 * NM * 4);

    // ---- A. timestep embedding ----
    k_time_embed<<<B, 256, 0, stream>>>(tsteps, sinEmb);
    k_small_linear<<<g1(B * D), 256, 0, stream>>>(sinEmb, t_w1, t_b1, temb1, B, 256, D, 2);
    k_small_linear<<<g1(B * D), 256, 0, stream>>>(temb1, t_w2, t_b2, temb, B, D, D, 0);
    k_silu<<<g1(B * D), 256, 0, stream>>>(temb, stemb, B * D);

    // ---- B. CSM fusion matrix M = P (sig(mask) * (P^T (ipw pdw^T + b) P)) P^T ----
    k_f2bf<<<g1(D * D), 256, 0, stream>>>(P, Pb, D * D);
    k_transpose_f2bf<<<g1(D * D), 256, 0, stream>>>(P, PTb, D, D);
    k_f2bf<<<g1(D * DL), 256, 0, stream>>>(ipw, ipwb, D * DL);
    k_f2bf<<<g1(D * DL), 256, 0, stream>>>(pdw, pdwb, D * DL);
    gemm(stream, ipwb,   pdwb, false, wprojb, true, D, D, DL, pdb, 0, nullptr, 0, 1, nullptr);
    gemm(stream, wprojb, PTb,  false, t1b,    true, D, D, D, nullptr, 0, nullptr, 0, 1, nullptr); // w_proj @ P
    gemm(stream, PTb,    t1b,  true,  wprimb, true, D, D, D, nullptr, 0, nullptr, 0, 1, nullptr); // P^T @ t1
    k_mask_sigmoid<<<g1(D * D), 256, 0, stream>>>(wprimb, smask, mwb, D * D);
    gemm(stream, Pb,     mwb,  true,  t2b,    true, D, D, D, nullptr, 0, nullptr, 0, 1, nullptr); // P @ mw
    gemm(stream, t2b,    Pb,   false, Mb,     true, D, D, D, nullptr, 0, nullptr, 0, 1, nullptr); // t2 @ P^T

    // ---- C. input projection + fused spectral mix + position conv ----
    k_f2bf<<<g1(M2 * DL), 256, 0, stream>>>(llama, xinb, M2 * DL);
    gemm(stream, xinb, ipwb, false, x0b, true, M2, D, DL, ipb, 0, nullptr, 0, 1, nullptr);
    gemm(stream, x0b,  Mb,   false, X,   false, M2, D, D, nullptr, 0, nullptr, 0, 1, nullptr);
    k_gconv<<<g1(M2 * D), 256, 0, stream>>>(X,  pc1_w, pc1_b, Y1, B, T, D, D / 16, 1);
    k_gconv<<<g1(M2 * D), 256, 0, stream>>>(Y1, pc2_w, pc2_b, X,  B, T, D, D / 16, 0);

    // ---- D. transformer blocks ----
    for (int blk = 0; blk < 8; ++blk) {
        const float* adaln_w = (const float*)d_in[BLK0 + blk * 16 + 0];
        const float* adaln_b = (const float*)d_in[BLK0 + blk * 16 + 1];
        const float* mod_w   = (const float*)d_in[BLK0 + blk * 16 + 2];
        const float* mod_b   = (const float*)d_in[BLK0 + blk * 16 + 3];
        const float* wq      = (const float*)d_in[BLK0 + blk * 16 + 4];
        const float* bq      = (const float*)d_in[BLK0 + blk * 16 + 5];
        const float* wk      = (const float*)d_in[BLK0 + blk * 16 + 6];
        const float* bk_     = (const float*)d_in[BLK0 + blk * 16 + 7];
        const float* wv      = (const float*)d_in[BLK0 + blk * 16 + 8];
        const float* bv      = (const float*)d_in[BLK0 + blk * 16 + 9];
        const float* wo      = (const float*)d_in[BLK0 + blk * 16 + 10];
        const float* bo      = (const float*)d_in[BLK0 + blk * 16 + 11];
        const float* ff_w1   = (const float*)d_in[BLK0 + blk * 16 + 12];
        const float* ff_b1   = (const float*)d_in[BLK0 + blk * 16 + 13];
        const float* ff_w2   = (const float*)d_in[BLK0 + blk * 16 + 14];
        const float* ff_b2   = (const float*)d_in[BLK0 + blk * 16 + 15];

        k_small_linear<<<g1(B * 3 * D), 256, 0, stream>>>(stemb, adaln_w, adaln_b, m3, B, D, 3 * D, 0);
        k_small_linear<<<g1(B * 6 * D), 256, 0, stream>>>(stemb, mod_w,   mod_b,   m6, B, D, 6 * D, 0);

        // x = ln(x)*(1+sc)+sh   (m3: sh@0, sc@D)  — in place
        k_ln<<<M2, 256, 0, stream>>>(X, D, m3, m3 + D, 3 * D, T, nullptr, nullptr, X, nullptr);
        // h = ln(x)*(1+ca)+sa   (m6: sa@0, ca@D) -> bf16
        k_ln<<<M2, 256, 0, stream>>>(X, D, m6, m6 + D, 6 * D, T, nullptr, nullptr, nullptr, hb);

        k_f2bf<<<g1(D * D), 256, 0, stream>>>(wq, wstage, D * D);
        gemm(stream, hb, wstage, false, qb, true, M2, D, D, bq, 0, nullptr, 0, 1, nullptr);
        k_f2bf<<<g1(D * D), 256, 0, stream>>>(wk, wstage, D * D);
        gemm(stream, hb, wstage, false, kb, true, M2, D, D, bk_, 0, nullptr, 0, 1, nullptr);
        k_f2bf<<<g1(D * D), 256, 0, stream>>>(wv, wstage, D * D);
        gemm(stream, hb, wstage, false, vb, true, M2, D, D, bv, 0, nullptr, 0, 1, nullptr);

        k_flash_attn<<<dim3(T / 64, H, B), 128, 0, stream>>>(qb, kb, vb, maskp, attnb, B, H, T, D);

        // x = x + ga * (attn @ wo^T + bo)    (ga @ 2D in m6)
        k_f2bf<<<g1(D * D), 256, 0, stream>>>(wo, wstage, D * D);
        gemm(stream, attnb, wstage, false, X, false, M2, D, D, bo, 0, m6 + 2 * D, 6 * D, T, X);

        // h = ln(x)*(1+cm)+sm  (sm @ 3D, cm @ 4D) -> bf16
        k_ln<<<M2, 256, 0, stream>>>(X, D, m6 + 3 * D, m6 + 4 * D, 6 * D, T, nullptr, nullptr, nullptr, hb);

        // ff = gelu(h @ w1^T + b1) ; x = x + gm * (ff @ w2^T + b2)   (gm @ 5D)
        k_f2bf<<<g1(FF * D), 256, 0, stream>>>(ff_w1, wstage, FF * D);
        gemm(stream, hb, wstage, false, fbuf, true, M2, FF, D, ff_b1, 1, nullptr, 0, 1, nullptr);
        k_f2bf<<<g1(FF * D), 256, 0, stream>>>(ff_w2, wstage, FF * D);
        gemm(stream, fbuf, wstage, false, X, false, M2, D, FF, ff_b2, 0, m6 + 5 * D, 6 * D, T, X);
    }

    // ---- E. final norm + mel head ----
    k_ln<<<M2, 256, 0, stream>>>(X, D, nullptr, nullptr, 0, T, fn_g, fn_b, nullptr, hb);
    k_f2bf<<<g1(NM * D), 256, 0, stream>>>(mel_w, wstage, NM * D);
    gemm(stream, hb, wstage, false, melTmp, false, M2, NM, D, mel_b, 0, nullptr, 0, 1, nullptr);
    k_mel_out<<<g1(B * NM * T), 256, 0, stream>>>(melTmp, (float*)d_out, B, T, NM);
}
